// KnowledgeGraphReasoner_81003083202651
// MI455X (gfx1250) — matrix operand order repaired and checked by reading, not compiled, not run on hardware
//
#include <hip/hip_runtime.h>

// ---------------------------------------------------------------------------
// KnowledgeGraphReasoner on MI455X (gfx1250):
//   scores = 2*(hr . E^T) - ||hr||^2 - ||E||^2 ; predictions = argmax
//
// Roofline: 52.4 GFLOP cross-term vs ~512 MB HBM traffic (410 MB of it the
// score matrix itself) -> need ~2.4 PFLOP/s effective to be HBM-bound at
// 23.3 TB/s. Only the WMMA pipe gets there: bf16 16x16x32 WMMA with fp32
// accumulation; the precision-critical norms stay fp32 end-to-end.
//
// GEMM structure (per 256-thread block = 8 wave32s):
//   block tile 64(M) x 128(N), waves 4(M) x 2(N), wave tile 16 x 64.
//   A (hr strip) hoisted: converted once to 8 resident v16bf fragments.
//   B (entity tile) staged cooperatively in LDS as bf16, double-buffered,
//   padded row stride (20 dwords) -> conflict-free b128 fragment reads.
//   K = 256 fully unrolled -> 32 v_wmma_f32_16x16x32_bf16 per wave.
// ---------------------------------------------------------------------------

#define N_ENTITIES 100000
#define N_RELATIONS 1000
#define EMBED_DIM 256
#define BATCH 1024

#define KSTEPS   (EMBED_DIM / 32)   // 8
#define BN       128                // block tile N
#define ROW_DW   16                 // 32 bf16 per staged row = 16 dwords
#define ROW_PAD  20                 // padded stride in dwords (bank spread)

typedef __attribute__((ext_vector_type(16))) __bf16 v16bf;
typedef __attribute__((ext_vector_type(8)))  float  v8f;

union FragCast { v16bf v; uint4 q[2]; };

// ---------------------------------------------------------------------------
// Prologue 1: hr[b][d] = E[head[b]][d] + R[rel[b]][d]; hr_sq[b] = ||hr_b||^2
// ---------------------------------------------------------------------------
__global__ __launch_bounds__(EMBED_DIM)
void kg_hr_kernel(const int* __restrict__ queries,
                  const float* __restrict__ ent,
                  const float* __restrict__ rel,
                  float* __restrict__ hr,
                  float* __restrict__ hr_sq) {
    const int b = blockIdx.x;
    const int d = threadIdx.x;
    const int head = queries[3 * b + 0];
    const int r    = queries[3 * b + 1];
    const float v = ent[(size_t)head * EMBED_DIM + d] + rel[(size_t)r * EMBED_DIM + d];
    hr[(size_t)b * EMBED_DIM + d] = v;

    __shared__ float red[EMBED_DIM];
    red[d] = v * v;
    __syncthreads();
    #pragma unroll
    for (int s = EMBED_DIM / 2; s > 0; s >>= 1) {
        if (d < s) red[d] += red[d + s];
        __syncthreads();
    }
    if (d == 0) hr_sq[b] = red[0];
}

// ---------------------------------------------------------------------------
// Prologue 2: ent_sq[n] = ||E_n||^2
// ---------------------------------------------------------------------------
__global__ __launch_bounds__(EMBED_DIM)
void kg_entsq_kernel(const float* __restrict__ ent, float* __restrict__ ent_sq) {
    const int n = blockIdx.x;
    const int d = threadIdx.x;
    const float v = ent[(size_t)n * EMBED_DIM + d];

    __shared__ float red[EMBED_DIM];
    red[d] = v * v;
    __syncthreads();
    #pragma unroll
    for (int s = EMBED_DIM / 2; s > 0; s >>= 1) {
        if (d < s) red[d] += red[d + s];
        __syncthreads();
    }
    if (d == 0) ent_sq[n] = red[0];
}

// ---------------------------------------------------------------------------
// Main kernel: bf16 WMMA GEMM + fused score epilogue.
// ---------------------------------------------------------------------------
__global__ __launch_bounds__(256)
void kg_score_wmma_kernel(const float* __restrict__ ent,
                          const float* __restrict__ hr,
                          const float* __restrict__ hr_sq,
                          const float* __restrict__ ent_sq,
                          float* __restrict__ out) {
    const int tid   = threadIdx.x;
    const int lane  = tid & 31;
    const int wave  = tid >> 5;
    const int waveM = wave & 3;   // 4 waves stacked in M
    const int waveN = wave >> 2;  // 2 waves stacked in N

    const int m0    = blockIdx.y * 64 + waveM * 16;   // batch-row base (in range)
    const int nblk  = blockIdx.x * BN;                // block entity-col base
    const int nbase = nblk + waveN * 64;              // wave entity-col base

    // Double-buffered bf16 B tile: [2][BN rows][ROW_PAD dwords]
    __shared__ __align__(16) unsigned int ldsB[2][BN * ROW_PAD];

    // ---- Hoisted A fragments: one wave strip of hr, fp32 -> bf16 once ----
    // 16-bit A(16x32): lanes 0-15 (rows) hold K 0..15,
    // lanes 16-31 hold K 16..31 (matches B fragment order).
    const int arow   = m0 + (lane & 15);
    const int khalfA = (lane >> 4) << 4;
    const float* arow_p = hr + (size_t)arow * EMBED_DIM;

    v16bf afr[KSTEPS];
    #pragma unroll
    for (int ks = 0; ks < KSTEPS; ++ks) {
        #pragma unroll
        for (int j = 0; j < 16; ++j) {
            const int k = j + khalfA;
            afr[ks][j] = (__bf16)arow_p[ks * 32 + k];
        }
    }

    // ---- B staging assignment: thread -> (col, k-half) of the 128x32 tile ----
    const int scol = tid >> 1;              // 0..127
    const int skh  = tid & 1;               // 0: K 0..15, 1: K 16..31
    int gcol = nblk + scol;
    gcol = (gcol < N_ENTITIES) ? gcol : (N_ENTITIES - 1);   // clamp: no branch
    const float* sbase = ent + (size_t)gcol * EMBED_DIM + skh * 16;
    const int sdst = scol * ROW_PAD + skh * 8;              // dword offset in stage

    // ---- B fragment read offsets (bf16 B 32x16: lanes 0-15 K 0..15, 16-31 K 16..31)
    const int fcol0 = waveN * 64 + (lane & 15);
    const int fkoff = (lane >> 4) << 3;     // +8 dwords for upper half lanes

    v8f acc[4] = {{}, {}, {}, {}};

    // ---- Preload stage 0 ----
    float4 r0, r1, r2, r3;
    {
        const float4* src = (const float4*)(sbase + 0);
        r0 = src[0]; r1 = src[1]; r2 = src[2]; r3 = src[3];
        FragCast u;
        #pragma unroll
        for (int j = 0; j < 4; ++j) { u.v[j]      = (__bf16)(&r0.x)[j]; }
        #pragma unroll
        for (int j = 0; j < 4; ++j) { u.v[4 + j]  = (__bf16)(&r1.x)[j]; }
        #pragma unroll
        for (int j = 0; j < 4; ++j) { u.v[8 + j]  = (__bf16)(&r2.x)[j]; }
        #pragma unroll
        for (int j = 0; j < 4; ++j) { u.v[12 + j] = (__bf16)(&r3.x)[j]; }
        uint4* dst = (uint4*)&ldsB[0][sdst];
        dst[0] = u.q[0]; dst[1] = u.q[1];
    }
    __syncthreads();

    // ---- Main K loop: fully unrolled, double buffered ----
    #pragma unroll
    for (int ks = 0; ks < KSTEPS; ++ks) {
        const int buf = ks & 1;

        // Issue next stage's global loads before consuming this stage.
        if (ks + 1 < KSTEPS) {
            const float4* src = (const float4*)(sbase + (ks + 1) * 32);
            r0 = src[0]; r1 = src[1]; r2 = src[2]; r3 = src[3];
        }

        // 4 WMMAs against the staged bf16 tile.
        #pragma unroll
        for (int i = 0; i < 4; ++i) {
            FragCast u;
            const uint4* p =
                (const uint4*)&ldsB[buf][(fcol0 + i * 16) * ROW_PAD + fkoff];
            u.q[0] = p[0]; u.q[1] = p[1];
            acc[i] = __builtin_amdgcn_wmma_f32_16x16x32_bf16(
                false, afr[ks], false, u.v, (short)0, acc[i], false, false);
        }

        // Convert + store next stage into the other buffer.
        if (ks + 1 < KSTEPS) {
            FragCast u;
            #pragma unroll
            for (int j = 0; j < 4; ++j) { u.v[j]      = (__bf16)(&r0.x)[j]; }
            #pragma unroll
            for (int j = 0; j < 4; ++j) { u.v[4 + j]  = (__bf16)(&r1.x)[j]; }
            #pragma unroll
            for (int j = 0; j < 4; ++j) { u.v[8 + j]  = (__bf16)(&r2.x)[j]; }
            #pragma unroll
            for (int j = 0; j < 4; ++j) { u.v[12 + j] = (__bf16)(&r3.x)[j]; }
            uint4* dst = (uint4*)&ldsB[buf ^ 1][sdst];
            dst[0] = u.q[0]; dst[1] = u.q[1];
        }
        __syncthreads();
    }

    // ---- Fused epilogue: score = 2*cross - hr_sq[m] - ent_sq[n] ----
    // C/D layout: VGPR r, lanes 0-15 -> M=r, N=lane; lanes 16-31 -> M=r+8.
    const int mhi = (lane >> 4) << 3;
    float hs[8];
    #pragma unroll
    for (int r = 0; r < 8; ++r) hs[r] = hr_sq[m0 + r + mhi];

    #pragma unroll
    for (int i = 0; i < 4; ++i) {
        const int n = nbase + i * 16 + (lane & 15);
        if (n < N_ENTITIES) {
            const float es = ent_sq[n];
            #pragma unroll
            for (int r = 0; r < 8; ++r) {
                const int m = m0 + r + mhi;
                out[(size_t)m * N_ENTITIES + n] = 2.0f * acc[i][r] - hs[r] - es;
            }
        }
    }
}

// ---------------------------------------------------------------------------
// Argmax per batch row; predictions stored as float after the score matrix.
// ---------------------------------------------------------------------------
__global__ __launch_bounds__(256)
void kg_argmax_kernel(const float* __restrict__ scores, float* __restrict__ pred) {
    const int b = blockIdx.x;
    const float* row = scores + (size_t)b * N_ENTITIES;

    float best = -__builtin_inff();
    int bidx = 0;
    for (int i = threadIdx.x; i < N_ENTITIES; i += blockDim.x) {
        const float v = row[i];
        if (v > best || (v == best && i < bidx)) { best = v; bidx = i; }
    }

    __shared__ float sv[256];
    __shared__ int   si[256];
    sv[threadIdx.x] = best;
    si[threadIdx.x] = bidx;
    __syncthreads();
    #pragma unroll
    for (int s = 128; s > 0; s >>= 1) {
        if (threadIdx.x < s) {
            const float ov = sv[threadIdx.x + s];
            const int   oi = si[threadIdx.x + s];
            if (ov > sv[threadIdx.x] ||
                (ov == sv[threadIdx.x] && oi < si[threadIdx.x])) {
                sv[threadIdx.x] = ov;
                si[threadIdx.x] = oi;
            }
        }
        __syncthreads();
    }
    if (threadIdx.x == 0) pred[b] = (float)si[0];
}

// ---------------------------------------------------------------------------
extern "C" void kernel_launch(void* const* d_in, const int* in_sizes, int n_in,
                              void* d_out, int out_size, void* d_ws, size_t ws_size,
                              hipStream_t stream) {
    (void)in_sizes; (void)n_in; (void)out_size; (void)ws_size;

    const int*   queries = (const int*)d_in[0];
    const float* ent     = (const float*)d_in[1];
    const float* rel     = (const float*)d_in[2];
    float*       out     = (float*)d_out;

    // Workspace layout (floats): hr[1024*256] | hr_sq[1024] | ent_sq[100000]  (~1.45 MB)
    float* ws     = (float*)d_ws;
    float* hr     = ws;
    float* hr_sq  = hr + (size_t)BATCH * EMBED_DIM;
    float* ent_sq = hr_sq + BATCH;

    kg_hr_kernel<<<BATCH, EMBED_DIM, 0, stream>>>(queries, ent, rel, hr, hr_sq);
    kg_entsq_kernel<<<N_ENTITIES, EMBED_DIM, 0, stream>>>(ent, ent_sq);

    dim3 grid((N_ENTITIES + BN - 1) / BN, BATCH / 64);  // 782 x 16
    kg_score_wmma_kernel<<<grid, 256, 0, stream>>>(ent, hr, hr_sq, ent_sq, out);

    kg_argmax_kernel<<<BATCH, 256, 0, stream>>>(out, out + (size_t)BATCH * N_ENTITIES);
}
